// SimpleRetrieverMemory_78718160601364
// MI455X (gfx1250) — matrix operand order repaired and checked by reading, not compiled, not run on hardware
//
#include <hip/hip_runtime.h>
#include <hip/hip_bf16.h>

// Problem constants (from reference)
#define BQ     2048      // queries
#define NK     131072    // keys
#define KD     256       // key/query dim
#define VD     256       // value dim
#define TOPK   8

// Tiling
#define NSPLIT 4                 // N-range split across blockIdx.x
#define BM     64                // query rows per workgroup
#define BN     32                // key columns per chunk iteration
#define KS     264               // padded LDS row stride in halves (264*2B -> bank-conflict-free b128)
#define ITERS  ((NK / NSPLIT) / BN)   // 1024

typedef __attribute__((ext_vector_type(16))) _Float16      v16h;
typedef __attribute__((ext_vector_type(8)))  float         v8f;
typedef __attribute__((ext_vector_type(4)))  unsigned int  v4u;

union Frag16 { v4u q[2]; v16h v; };          // 32B WMMA operand (16 x f16)
union Pack8  { _Float16 h[8]; v4u q; };      // 8 f32 -> 8 f16 packed store

#define FLT_BIG 3.4e38f

// ---------------------------------------------------------------------------
// Kernel 1: k2[n] = ||keys[n]||^2   (fp32, wave32 shuffle reduction)
// ---------------------------------------------------------------------------
__global__ __launch_bounds__(256) void k2_kernel(const float* __restrict__ keys,
                                                 float* __restrict__ k2) {
    const int wave = threadIdx.x >> 5;
    const int lane = threadIdx.x & 31;
    const int row  = blockIdx.x * 8 + wave;           // one row per wave
    const float4* p = (const float4*)(keys + (size_t)row * KD);
    float s = 0.0f;
#pragma unroll
    for (int pass = 0; pass < 2; ++pass) {            // 2 * 32 lanes * float4 = 256
        float4 v = p[pass * 32 + lane];
        s = fmaf(v.x, v.x, s); s = fmaf(v.y, v.y, s);
        s = fmaf(v.z, v.z, s); s = fmaf(v.w, v.w, s);
    }
#pragma unroll
    for (int off = 16; off; off >>= 1) s += __shfl_xor(s, off, 32);
    if (lane == 0) k2[row] = s;
}

// ---------------------------------------------------------------------------
// Kernel 2: WMMA scoring + streaming per-thread top-8 + per-row merge.
// Grid: (NSPLIT, BQ/BM). Block: 256 threads = 8 waves (4 M-subtiles x 2 N-subtiles).
// ---------------------------------------------------------------------------
__global__ __launch_bounds__(256) void score_topk_kernel(
        const float* __restrict__ query, const float* __restrict__ keys,
        const float* __restrict__ k2,
        float* __restrict__ cand_s, unsigned int* __restrict__ cand_i) {

    __shared__ __align__(16) _Float16 lds_q[BM * KS];   // queries f16 (33792 B)
    __shared__ __align__(16) _Float16 lds_k[BN * KS];   // key chunk f16 (16896 B)
    __shared__ __align__(16) float    lds_s[BM * 36];   // score tile, padded (9216 B)

    const int tid   = threadIdx.x;
    const int lane  = tid & 31;
    const int wave  = tid >> 5;
    const int cl    = lane & 15;       // column-in-subtile / row-in-subtile
    const int g     = lane >> 4;       // lane half-group
    const int m0    = (wave & 3) * 16; // M-subtile base
    const int c0    = (wave >> 2) * 16;// N-subtile base
    const int b0    = blockIdx.y * BM;
    const int split = blockIdx.x;
    const int nsbase = split * (NK / NSPLIT);

    // ---- Stage queries fp32 -> f16 into LDS, once ----
    for (int e = tid * 8; e < BM * KD; e += 256 * 8) {
        const int m = e >> 8, k = e & 255;
        const float4* src = (const float4*)(query + (size_t)(b0 + m) * KD + k);
        float4 f0 = src[0], f1 = src[1];
        Pack8 pk;
        pk.h[0] = (_Float16)f0.x; pk.h[1] = (_Float16)f0.y;
        pk.h[2] = (_Float16)f0.z; pk.h[3] = (_Float16)f0.w;
        pk.h[4] = (_Float16)f1.x; pk.h[5] = (_Float16)f1.y;
        pk.h[6] = (_Float16)f1.z; pk.h[7] = (_Float16)f1.w;
        *(v4u*)&lds_q[m * KS + k] = pk.q;
    }

    // Per-thread sorted top-8 (ascending; best[7] is the current threshold)
    float        best[TOPK];
    unsigned int bidx[TOPK];
#pragma unroll
    for (int j = 0; j < TOPK; ++j) { best[j] = FLT_BIG; bidx[j] = 0u; }

    const int scan_row = tid >> 2;       // 64 rows, 4 threads per row
    const int scan_q   = tid & 3;        // each scans 8 of 32 columns

    __syncthreads();

    for (int it = 0; it < ITERS; ++it) {
        const int n_base = nsbase + it * BN;

        // ---- Stage 32 key rows fp32 -> f16 into LDS ----
        {
            const int c   = tid >> 3;    // 0..31 key column
            const int seg = tid & 7;     // 0..7, 32 dims each
            const float* gk = keys + (size_t)(n_base + c) * KD + seg * 32;
            if (it + 1 < ITERS) __builtin_prefetch(gk + (size_t)BN * KD, 0, 1);
            _Float16* dst = &lds_k[c * KS + seg * 32];
#pragma unroll
            for (int u = 0; u < 4; ++u) {
                float4 f0 = ((const float4*)gk)[u * 2 + 0];
                float4 f1 = ((const float4*)gk)[u * 2 + 1];
                Pack8 pk;
                pk.h[0] = (_Float16)f0.x; pk.h[1] = (_Float16)f0.y;
                pk.h[2] = (_Float16)f0.z; pk.h[3] = (_Float16)f0.w;
                pk.h[4] = (_Float16)f1.x; pk.h[5] = (_Float16)f1.y;
                pk.h[6] = (_Float16)f1.z; pk.h[7] = (_Float16)f1.w;
                *(v4u*)(dst + u * 8) = pk.q;
            }
        }
        __syncthreads();   // lds_k ready; previous scan done -> lds_s writable

        // ---- qk tile via WMMA: 8 k-steps of 16x16x32 f16, f32 accumulate ----
        {
            v8f acc = {};
            const _Float16* qrow = lds_q + (m0 + cl) * KS;
            const _Float16* krow = lds_k + (c0 + cl) * KS;
#pragma unroll
            for (int kk = 0; kk < 8; ++kk) {
                Frag16 A, Bf;
                const int ka = kk * 32 + g * 8;   // A: lanes 0-15 K{0-7,16-23}, lanes 16-31 K{8-15,24-31}
                A.q[0] = *(const v4u*)(qrow + ka);
                A.q[1] = *(const v4u*)(qrow + ka + 16);
                const int kb = kk * 32 + g * 8;   // B: lanes 0-15 K{0-7,16-23}, lanes 16-31 K{8-15,24-31}
                Bf.q[0] = *(const v4u*)(krow + kb);
                Bf.q[1] = *(const v4u*)(krow + kb + 16);
                acc = __builtin_amdgcn_wmma_f32_16x16x32_f16(
                        false, A.v, false, Bf.v, (short)0, acc, false, false);
            }
            // score = k2 - 2*qk (q2 dropped: constant per row, sqrt monotone)
            const float k2v = k2[n_base + c0 + cl];
#pragma unroll
            for (int r = 0; r < 8; ++r) {
                const int row = m0 + r + g * 8;   // D layout: VGPR r -> M=r (lanes<16) / M=r+8
                lds_s[row * 36 + c0 + cl] = fmaf(-2.0f, acc[r], k2v);
            }
        }
        __syncthreads();   // scores visible

        // ---- Streaming top-8 scan (register-resident sorted insert) ----
        {
            const float* srow = &lds_s[scan_row * 36 + scan_q * 8];
            const unsigned int nb = (unsigned int)(n_base + scan_q * 8);
#pragma unroll
            for (int j = 0; j < 8; ++j) {
                const float s = srow[j];
                if (s < best[TOPK - 1]) {
                    float cs_ = s; unsigned int ci_ = nb + j;
#pragma unroll
                    for (int p = 0; p < TOPK; ++p) {
                        if (cs_ < best[p]) {
                            float tf = best[p]; unsigned int ti = bidx[p];
                            best[p] = cs_; bidx[p] = ci_;
                            cs_ = tf; ci_ = ti;
                        }
                    }
                }
            }
        }
    }

    // ---- Per-row merge: 4 threads x 8 candidates -> top-8 of 32 ----
    __syncthreads();                              // lds_q no longer needed; reuse
    float*        cs = (float*)lds_q;             // 64*32 floats (8 KB)
    unsigned int* ci = (unsigned int*)(lds_q + 8192); // 8 KB at byte offset 16 KB
#pragma unroll
    for (int j = 0; j < TOPK; ++j) {
        cs[scan_row * 32 + scan_q * 8 + j] = best[j];
        ci[scan_row * 32 + scan_q * 8 + j] = bidx[j];
    }
    __syncthreads();
    if (tid < BM) {
        const int r = tid;
        const size_t base = ((size_t)(b0 + r) * NSPLIT + split) * TOPK;
        for (int s = 0; s < TOPK; ++s) {
            float bv = FLT_BIG; unsigned int bi = 0xFFFFFFFFu; int bj = 0;
            for (int j = 0; j < 32; ++j) {
                const float v = cs[r * 32 + j];
                const unsigned int id = ci[r * 32 + j];
                if (v < bv || (v == bv && id < bi)) { bv = v; bi = id; bj = j; }
            }
            cs[r * 32 + bj] = FLT_BIG;            // remove winner
            cand_s[base + s] = bv;
            cand_i[base + s] = bi;
        }
    }
}

// ---------------------------------------------------------------------------
// Kernel 3: merge NSPLIT*8 candidates per query -> final top-8, gather vals.
// ---------------------------------------------------------------------------
__global__ __launch_bounds__(256) void merge_gather_kernel(
        const float* __restrict__ vals,
        const float* __restrict__ cand_s, const unsigned int* __restrict__ cand_i,
        float* __restrict__ out) {
    __shared__ float        ms[NSPLIT * TOPK];
    __shared__ unsigned int mi[NSPLIT * TOPK];
    __shared__ unsigned int sel[TOPK];
    const int b   = blockIdx.x;
    const int tid = threadIdx.x;
    if (tid < NSPLIT * TOPK) {
        ms[tid] = cand_s[(size_t)b * NSPLIT * TOPK + tid];
        mi[tid] = cand_i[(size_t)b * NSPLIT * TOPK + tid];
    }
    __syncthreads();
    if (tid == 0) {
        for (int s = 0; s < TOPK; ++s) {          // stable: tie -> lower index
            float bv = FLT_BIG; unsigned int bi = 0xFFFFFFFFu; int bj = 0;
            for (int j = 0; j < NSPLIT * TOPK; ++j) {
                const float v = ms[j];
                if (v < bv || (v == bv && mi[j] < bi)) { bv = v; bi = mi[j]; bj = j; }
            }
            ms[bj] = FLT_BIG;
            sel[s] = bi;
        }
    }
    __syncthreads();
    // Gather vals rows -> out[b][s][:], coalesced over v
    for (int e = tid; e < TOPK * VD; e += 256) {
        const int s = e >> 8, v = e & 255;
        out[(size_t)b * (TOPK * VD) + e] = vals[(size_t)sel[s] * VD + v];
    }
}

// ---------------------------------------------------------------------------
extern "C" void kernel_launch(void* const* d_in, const int* in_sizes, int n_in,
                              void* d_out, int out_size, void* d_ws, size_t ws_size,
                              hipStream_t stream) {
    const float* query = (const float*)d_in[0];   // [B, KD]
    const float* keys  = (const float*)d_in[1];   // [N, KD]
    const float* vals  = (const float*)d_in[2];   // [N, VD]
    // d_in[3] = top_k scalar (== 8, compile-time constant here)
    float* out = (float*)d_out;

    // Workspace layout: k2[N] | cand_s[B*NSPLIT*8] | cand_i[B*NSPLIT*8]  (~1 MB)
    float*        k2     = (float*)d_ws;
    float*        cand_s = (float*)((char*)d_ws + (size_t)NK * 4);
    unsigned int* cand_i = (unsigned int*)((char*)d_ws + (size_t)NK * 4 +
                                           (size_t)BQ * NSPLIT * TOPK * 4);

    k2_kernel<<<NK / 8, 256, 0, stream>>>(keys, k2);
    score_topk_kernel<<<dim3(NSPLIT, BQ / BM), 256, 0, stream>>>(
        query, keys, k2, cand_s, cand_i);
    merge_gather_kernel<<<BQ, 256, 0, stream>>>(vals, cand_s, cand_i, out);
}
